// EdgeEarlyInteraction1Baseline_16716012716574
// MI455X (gfx1250) — matrix operand order repaired and checked by reading, not compiled, not run on hardware
//
#include <hip/hip_runtime.h>
#include <hip/hip_bf16.h>

// ---------------------------------------------------------------------------
// EdgeEarlyInteraction (ISONET-style) for MI455X / gfx1250.
// All dense algebra runs through v_wmma_f32_16x16x32_f16 (wave32 WMMA),
// 2x2 register-blocked (4 WMMAs / k-step), all operands pre-transposed so
// every fragment load is a contiguous global_load_b128 pair.
// ---------------------------------------------------------------------------

typedef _Float16 half_t;
typedef _Float16 v16h __attribute__((ext_vector_type(16)));
typedef float    v8f  __attribute__((ext_vector_type(8)));

namespace {
constexpr int  kB   = 256;            // graph pairs
constexpr int  kM   = 128;            // padded edge set size
constexpr int  kEG  = 96;             // edges per graph
constexpr int  kNG  = 48;             // nodes per graph
constexpr int  kD   = 128;            // state dim
constexpr int  kSFD = 64;             // sinkhorn feature dim
constexpr int  kG   = 2 * kB;         // 512 graphs
constexpr int  kN   = kG * kNG;       // 24576 nodes
constexpr int  kE   = kG * kEG;       // 49152 edges
constexpr int  kGM  = kG * kM;        // 65536 padded edge rows
constexpr float kInvTemp = 10.0f;     // 1 / TEMP
}

// --- fragment loaders -------------------------------------------------------
// A (16x32 f16, MxK): lane L<16 -> row L, K = {kb+0..7, kb+16..23};
//                     lane L>=16 -> row L-16, K = {kb+8..15, kb+24..31}.
__device__ __forceinline__ v16h fragA(const half_t* __restrict__ p) {
  v16h r;
#pragma unroll
  for (int i = 0; i < 8; ++i) { r[i] = p[i]; r[i + 8] = p[i + 16]; }
  return r;
}
// B (32x16 f16, KxN) given as [N,K] rows: lane L<16 -> col L, K = {kb+0..7, kb+16..23};
//                                          lane L>=16 -> col L-16, K = {kb+8..15, kb+24..31}.
__device__ __forceinline__ v16h fragB(const half_t* __restrict__ p) {
  v16h r;
#pragma unroll
  for (int i = 0; i < 8; ++i) { r[i] = p[i]; r[i + 8] = p[i + 16]; }
  return r;
}

__global__ void wmma_gemm_kernel(const half_t* __restrict__ A,
                                 const half_t* __restrict__ B,
                                 const float* __restrict__ bias,
                                 float* __restrict__ Cf,
                                 half_t* __restrict__ Ch,
                                 int Mt2, int Nt2, int K,
                                 int lda, int ldb, int ldc,
                                 long long sA, long long sB, long long sC,
                                 int relu, int maskmod, float alpha) {
  const int lane = threadIdx.x & 31;
  const int wave = threadIdx.x >> 5;
  const int tile = blockIdx.x * 4 + wave;          // wave-uniform
  if (tile >= Mt2 * Nt2) return;                   // whole-wave exit only
  const int mt = tile / Nt2, nt = tile % Nt2;      // 32x32 output block

  const half_t* Ab = A + (long long)blockIdx.z * sA;
  const half_t* Bb = B + (long long)blockIdx.z * sB;
  const long long boffC = (long long)blockIdx.z * sC;

  const int r       = lane & 15;
  const int koff8   = (lane >> 4) << 3;   // A interleave
  const int koff16  = (lane >> 4) << 3;   // B interleave

  const half_t* a0p = Ab + (long long)(mt * 32 + r) * lda + koff8;
  const half_t* a1p = a0p + 16LL * lda;
  const half_t* b0p = Bb + (long long)(nt * 32 + r) * ldb + koff16;
  const half_t* b1p = b0p + 16LL * ldb;

  v8f acc00 = {}, acc01 = {}, acc10 = {}, acc11 = {};
  for (int kb = 0; kb < K; kb += 32) {
    __builtin_prefetch(a0p + kb + 32, 0, 1);
    __builtin_prefetch(b0p + kb + 32, 0, 1);
    v16h a0 = fragA(a0p + kb);
    v16h a1 = fragA(a1p + kb);
    v16h b0 = fragB(b0p + kb);
    v16h b1 = fragB(b1p + kb);
    acc00 = __builtin_amdgcn_wmma_f32_16x16x32_f16(false, a0, false, b0, (short)0, acc00, false, false);
    acc01 = __builtin_amdgcn_wmma_f32_16x16x32_f16(false, a0, false, b1, (short)0, acc01, false, false);
    acc10 = __builtin_amdgcn_wmma_f32_16x16x32_f16(false, a1, false, b0, (short)0, acc10, false, false);
    acc11 = __builtin_amdgcn_wmma_f32_16x16x32_f16(false, a1, false, b1, (short)0, acc11, false, false);
  }

  const int rowb = mt * 32 + ((lane >> 4) << 3);   // C/D: VGPR v -> M = v (+8 hi half)
  const int colb = nt * 32 + r;                    // N = lane&15
  const float bv0 = bias ? bias[colb]      : 0.0f;
  const float bv1 = bias ? bias[colb + 16] : 0.0f;
#pragma unroll
  for (int ti = 0; ti < 2; ++ti) {
    v8f aL = ti ? acc10 : acc00;
    v8f aR = ti ? acc11 : acc01;
#pragma unroll
    for (int i = 0; i < 8; ++i) {
      const int row = rowb + ti * 16 + i;
      float v0 = aL[i] * alpha + bv0;
      float v1 = aR[i] * alpha + bv1;
      if (relu) { v0 = fmaxf(v0, 0.0f); v1 = fmaxf(v1, 0.0f); }
      if (maskmod && ((row & (kM - 1)) >= kEG)) { v0 = 0.0f; v1 = 0.0f; }
      const long long o = boffC + (long long)row * ldc + colb;
      if (Cf) { Cf[o] = v0; Cf[o + 16] = v1; }
      if (Ch) { Ch[o] = (half_t)v0; Ch[o + 16] = (half_t)v1; }
    }
  }
}

// --- elementwise / glue kernels --------------------------------------------
__global__ void k_convert_wT(const float* __restrict__ W, half_t* __restrict__ Wt,
                             int K, int N, int Kpad) {
  long long idx = (long long)blockIdx.x * blockDim.x + threadIdx.x;
  long long total = (long long)N * Kpad;
  if (idx >= total) return;
  int n = (int)(idx / Kpad), k = (int)(idx % Kpad);
  Wt[idx] = (k < K) ? (half_t)W[(long long)k * N + n] : (half_t)0.0f;
}

__global__ void k_convert_pad(const float* __restrict__ X, half_t* __restrict__ Y,
                              int R, int C, int Cpad) {
  long long idx = (long long)blockIdx.x * blockDim.x + threadIdx.x;
  long long total = (long long)R * Cpad;
  if (idx >= total) return;
  int r = (int)(idx / Cpad), c = (int)(idx % Cpad);
  Y[idx] = (c < C) ? (half_t)X[(long long)r * C + c] : (half_t)0.0f;
}

// pad edges into [G, M, D] and also write transposed [G, D, M] copy so the
// alignment GEMMs (P @ sc, P^T @ sq) get a fast [N,K] operand.
__global__ void k_pad_edges(const half_t* __restrict__ eh,
                            half_t* __restrict__ ep, half_t* __restrict__ epT) {
  long long idx = (long long)blockIdx.x * blockDim.x + threadIdx.x;
  long long total = (long long)kGM * kD;
  if (idx >= total) return;
  int row = (int)(idx >> 7), d = (int)(idx & 127);
  int g = row >> 7, e = row & 127;
  half_t v = (e < kEG) ? eh[((long long)(g * kEG + e) << 7) + d] : (half_t)0.0f;
  ep[idx] = v;
  epT[(((long long)g * kD + d) << 7) + e] = v;
}

__global__ void k_sink_row(float* __restrict__ la) {
  __shared__ float sm[128];
  float* row = la + ((long long)blockIdx.x << 7);
  int t = threadIdx.x;
  float v = row[t];
  sm[t] = v; __syncthreads();
  for (int s = 64; s > 0; s >>= 1) { if (t < s) sm[t] = fmaxf(sm[t], sm[t + s]); __syncthreads(); }
  float mx = sm[0]; __syncthreads();
  sm[t] = expf(v - mx); __syncthreads();
  for (int s = 64; s > 0; s >>= 1) { if (t < s) sm[t] += sm[t + s]; __syncthreads(); }
  row[t] = v - (mx + logf(sm[0]));
}

__global__ void k_sink_col(float* __restrict__ la) {
  __shared__ float sm[128];
  int b = blockIdx.x >> 7, j = blockIdx.x & 127;
  float* base = la + ((long long)b << 14) + j;
  int t = threadIdx.x;
  float v = base[(long long)t << 7];
  sm[t] = v; __syncthreads();
  for (int s = 64; s > 0; s >>= 1) { if (t < s) sm[t] = fmaxf(sm[t], sm[t + s]); __syncthreads(); }
  float mx = sm[0]; __syncthreads();
  sm[t] = expf(v - mx); __syncthreads();
  for (int s = 64; s > 0; s >>= 1) { if (t < s) sm[t] += sm[t + s]; __syncthreads(); }
  base[(long long)t << 7] = v - (mx + logf(sm[0]));
}

__global__ void k_sink_fin(const float* __restrict__ la,
                           half_t* __restrict__ P, half_t* __restrict__ PT) {
  long long idx = (long long)blockIdx.x * blockDim.x + threadIdx.x;
  long long total = (long long)kB * kM * kM;
  if (idx >= total) return;
  int b = (int)(idx >> 14), ij = (int)(idx & 16383);
  int i = ij >> 7, j = ij & 127;
  float p = expf(la[idx]);
  P[idx] = (half_t)p;
  PT[((long long)b << 14) + ((long long)j << 7) + i] = (half_t)p;
}

__global__ void k_cat2(const half_t* __restrict__ eh, const half_t* __restrict__ qc,
                       const half_t* __restrict__ cq, half_t* __restrict__ out) {
  long long idx = (long long)blockIdx.x * blockDim.x + threadIdx.x;
  long long total = (long long)kE * 256;
  if (idx >= total) return;
  int row = (int)(idx >> 8), c = (int)(idx & 255);
  half_t v;
  if (c < 128) {
    v = eh[((long long)row << 7) + c];
  } else {
    int g = row / kEG, e = row - g * kEG, b = g >> 1;
    const half_t* src = (g & 1) ? cq : qc;
    v = src[((long long)b << 14) + ((long long)e << 7) + (c - 128)];
  }
  out[idx] = v;
}

__global__ void k_msgin(const half_t* __restrict__ nh, const int* __restrict__ fi,
                        const int* __restrict__ ti, const half_t* __restrict__ comb,
                        half_t* __restrict__ out) {
  long long idx = (long long)blockIdx.x * blockDim.x + threadIdx.x;
  long long total = (long long)kE * 384;
  if (idx >= total) return;
  int row = (int)(idx / 384), c = (int)(idx % 384);
  half_t v;
  if (c < 128)       v = nh[((long long)fi[row] << 7) + c];
  else if (c < 256)  v = nh[((long long)ti[row] << 7) + (c - 128)];
  else               v = comb[((long long)row << 7) + (c - 256)];
  out[idx] = v;
}

__global__ void k_updin(const half_t* __restrict__ nh, const float* __restrict__ agg,
                        half_t* __restrict__ out) {
  long long idx = (long long)blockIdx.x * blockDim.x + threadIdx.x;
  long long total = (long long)kN * 256;
  if (idx >= total) return;
  int row = (int)(idx >> 8), c = (int)(idx & 255);
  out[idx] = (c < 128) ? nh[((long long)row << 7) + c]
                       : (half_t)agg[((long long)row << 7) + (c - 128)];
}

__global__ void k_zero(float* __restrict__ p, long long n) {
  long long idx = (long long)blockIdx.x * blockDim.x + threadIdx.x;
  if (idx < n) p[idx] = 0.0f;
}

__global__ void k_scatter(const float* __restrict__ msgs, const int* __restrict__ ti,
                          float* __restrict__ agg) {
  long long idx = (long long)blockIdx.x * blockDim.x + threadIdx.x;
  long long total = (long long)kE * kD;
  if (idx >= total) return;
  int e = (int)(idx >> 7), d = (int)(idx & 127);
  atomicAdd(&agg[((long long)ti[e] << 7) + d], msgs[idx]);
}

__global__ void k_score(const half_t* __restrict__ ep, const float* __restrict__ qcf,
                        float* __restrict__ out) {
  __shared__ float sm[256];
  int b = blockIdx.x, t = threadIdx.x;
  const half_t* sq = ep + ((long long)b << 15);        // query graph 2b, [128,128]
  const float* r   = qcf + ((long long)b << 14);
  float s = 0.0f;
  for (int i = t; i < kM * kD; i += 256)
    s += fmaxf((float)sq[i] - r[i], 0.0f);
  sm[t] = s; __syncthreads();
  for (int st = 128; st > 0; st >>= 1) { if (t < st) sm[t] += sm[t + st]; __syncthreads(); }
  if (t == 0) out[b] = -sm[0];
}

// --- host ------------------------------------------------------------------
static inline unsigned ew_blocks(long long total) {
  return (unsigned)((total + 255) / 256);
}

extern "C" void kernel_launch(void* const* d_in, const int* in_sizes, int n_in,
                              void* d_out, int out_size, void* d_ws, size_t ws_size,
                              hipStream_t stream) {
  (void)in_sizes; (void)n_in; (void)out_size; (void)ws_size;
  const float* node_features = (const float*)d_in[0];
  const float* edge_features = (const float*)d_in[1];
  const int*   from_idx      = (const int*)d_in[2];
  const int*   to_idx        = (const int*)d_in[3];
  const float* enc_node_w = (const float*)d_in[4];  const float* enc_node_b = (const float*)d_in[5];
  const float* enc_edge_w = (const float*)d_in[6];  const float* enc_edge_b = (const float*)d_in[7];
  const float* msg_w1 = (const float*)d_in[8];      const float* msg_b1 = (const float*)d_in[9];
  const float* msg_w2 = (const float*)d_in[10];     const float* msg_b2 = (const float*)d_in[11];
  const float* upd_w1 = (const float*)d_in[12];     const float* upd_b1 = (const float*)d_in[13];
  const float* upd_w2 = (const float*)d_in[14];     const float* upd_b2 = (const float*)d_in[15];
  const float* int_w1 = (const float*)d_in[16];     const float* int_b1 = (const float*)d_in[17];
  const float* int_w2 = (const float*)d_in[18];     const float* int_b2 = (const float*)d_in[19];
  const float* sk_w1  = (const float*)d_in[20];     const float* sk_b1  = (const float*)d_in[21];
  const float* sk_w2  = (const float*)d_in[22];     const float* sk_b2  = (const float*)d_in[23];
  float* out = (float*)d_out;

  char* ws = (char*)d_ws;
  size_t off = 0;
  auto alloc = [&](size_t bytes) -> void* {
    void* p = ws + off; off += (bytes + 255) & ~(size_t)255; return p;
  };

  half_t* nf_h   = (half_t*)alloc((size_t)kN * 32 * 2);
  half_t* ef_h   = (half_t*)alloc((size_t)kE * 32 * 2);
  half_t* node_h = (half_t*)alloc((size_t)kN * kD * 2);
  half_t* edge_h = (half_t*)alloc((size_t)kE * kD * 2);
  half_t* ep     = (half_t*)alloc((size_t)kGM * kD * 2);
  half_t* epT    = (half_t*)alloc((size_t)kG * kD * kM * 2);
  half_t* skh1   = (half_t*)alloc((size_t)kGM * kSFD * 2);
  half_t* skf    = (half_t*)alloc((size_t)kGM * kSFD * 2);
  float*  la     = (float*)alloc((size_t)kB * kM * kM * 4);   // also reused as final P@sc
  half_t* P_h    = (half_t*)alloc((size_t)kB * kM * kM * 2);
  half_t* PT_h   = (half_t*)alloc((size_t)kB * kM * kM * 2);
  half_t* qc_h   = (half_t*)alloc((size_t)kB * kM * kD * 2);
  half_t* cq_h   = (half_t*)alloc((size_t)kB * kM * kD * 2);
  half_t* cat2   = (half_t*)alloc((size_t)kE * 256 * 2);
  half_t* h256   = (half_t*)alloc((size_t)kE * 256 * 2);
  half_t* comb_h = (half_t*)alloc((size_t)kE * kD * 2);
  half_t* msgin  = (half_t*)alloc((size_t)kE * 384 * 2);
  float*  msgs   = (float*)alloc((size_t)kE * kD * 4);
  float*  agg    = (float*)alloc((size_t)kN * kD * 4);
  half_t* updin  = (half_t*)alloc((size_t)kN * 256 * 2);
  half_t* wt_encn = (half_t*)alloc((size_t)kD * 32 * 2);
  half_t* wt_ence = (half_t*)alloc((size_t)kD * 32 * 2);
  half_t* wt_msg1 = (half_t*)alloc((size_t)256 * 384 * 2);
  half_t* wt_msg2 = (half_t*)alloc((size_t)kD * 256 * 2);
  half_t* wt_upd1 = (half_t*)alloc((size_t)256 * 256 * 2);
  half_t* wt_upd2 = (half_t*)alloc((size_t)kD * 256 * 2);
  half_t* wt_int1 = (half_t*)alloc((size_t)256 * 256 * 2);
  half_t* wt_int2 = (half_t*)alloc((size_t)kD * 256 * 2);
  half_t* wt_sk1  = (half_t*)alloc((size_t)kSFD * kD * 2);
  half_t* wt_sk2  = (half_t*)alloc((size_t)kSFD * kSFD * 2);

  auto cvtW = [&](const float* W, half_t* Wt, int K, int N, int Kpad) {
    k_convert_wT<<<ew_blocks((long long)N * Kpad), 256, 0, stream>>>(W, Wt, K, N, Kpad);
  };
  // C[Mr x Nc] = act(alpha * A[Mr x K] @ Bt[Nc x K]^T + bias)
  auto gemm = [&](const half_t* A, const half_t* Bt, const float* bias,
                  float* Cf, half_t* Ch, int Mr, int Nc, int K,
                  int lda, int ldb, int ldc,
                  long long sA, long long sB, long long sC, int batches,
                  int relu, int maskmod, float alpha) {
    int Mt2 = Mr / 32, Nt2 = Nc / 32;
    dim3 grid((unsigned)(((long long)Mt2 * Nt2 + 3) / 4), 1, (unsigned)batches);
    wmma_gemm_kernel<<<grid, 128, 0, stream>>>(A, Bt, bias, Cf, Ch, Mt2, Nt2, K,
        lda, ldb, ldc, sA, sB, sC, relu, maskmod, alpha);
  };

  // weights -> transposed f16
  cvtW(enc_node_w, wt_encn, 32, kD, 32);
  cvtW(enc_edge_w, wt_ence, 16, kD, 32);
  cvtW(msg_w1, wt_msg1, 384, 256, 384);
  cvtW(msg_w2, wt_msg2, 256, kD, 256);
  cvtW(upd_w1, wt_upd1, 256, 256, 256);
  cvtW(upd_w2, wt_upd2, 256, kD, 256);
  cvtW(int_w1, wt_int1, 256, 256, 256);
  cvtW(int_w2, wt_int2, 256, kD, 256);
  cvtW(sk_w1, wt_sk1, kD, kSFD, kD);
  cvtW(sk_w2, wt_sk2, kSFD, kSFD, kSFD);

  // features -> f16 (edge K padded 16->32)
  k_convert_pad<<<ew_blocks((long long)kN * 32), 256, 0, stream>>>(node_features, nf_h, kN, 32, 32);
  k_convert_pad<<<ew_blocks((long long)kE * 32), 256, 0, stream>>>(edge_features, ef_h, kE, 16, 32);

  // encoders
  gemm(nf_h, wt_encn, enc_node_b, nullptr, node_h, kN, kD, 32, 32, 32, kD, 0, 0, 0, 1, 0, 0, 1.0f);
  gemm(ef_h, wt_ence, enc_edge_b, nullptr, edge_h, kE, kD, 32, 32, 32, kD, 0, 0, 0, 1, 0, 0, 1.0f);

  for (int step = 0; step < 3; ++step) {
    const bool fin = (step == 2);
    k_pad_edges<<<ew_blocks((long long)kGM * kD), 256, 0, stream>>>(edge_h, ep, epT);
    // sinkhorn features: mlp2 then pad-row mask
    gemm(ep, wt_sk1, sk_b1, nullptr, skh1, kGM, kSFD, kD, kD, kD, kSFD, 0, 0, 0, 1, 1, 0, 1.0f);
    gemm(skh1, wt_sk2, sk_b2, nullptr, skf, kGM, kSFD, kSFD, kSFD, kSFD, kSFD, 0, 0, 0, 1, 0, 1, 1.0f);
    // log_alpha/TEMP = (fq @ fc^T) * 10   (fc rows already are the [N,K] operand)
    gemm(skf, skf + (size_t)kM * kSFD, nullptr, la, nullptr, kM, kM, kSFD,
         kSFD, kSFD, kM, 2LL * kM * kSFD, 2LL * kM * kSFD, (long long)kM * kM, kB,
         0, 0, kInvTemp);
    for (int it = 0; it < 8; ++it) {
      k_sink_row<<<kB * kM, kM, 0, stream>>>(la);
      k_sink_col<<<kB * kM, kM, 0, stream>>>(la);
    }
    k_sink_fin<<<ew_blocks((long long)kB * kM * kM), 256, 0, stream>>>(la, P_h, PT_h);

    if (!fin) {
      // qc = P @ sc   (Bt = sc^T = epT of odd graphs)
      gemm(P_h, epT + (size_t)kD * kM, nullptr, nullptr, qc_h, kM, kD, kM,
           kM, kM, kD, (long long)kM * kM, 2LL * kD * kM, (long long)kM * kD, kB, 0, 0, 1.0f);
      // cq = P^T @ sq (Bt = sq^T = epT of even graphs)
      gemm(PT_h, epT, nullptr, nullptr, cq_h, kM, kD, kM,
           kM, kM, kD, (long long)kM * kM, 2LL * kD * kM, (long long)kM * kD, kB, 0, 0, 1.0f);
      k_cat2<<<ew_blocks((long long)kE * 256), 256, 0, stream>>>(edge_h, qc_h, cq_h, cat2);
      // combined = mlp2([edge_enc, inter])
      gemm(cat2, wt_int1, int_b1, nullptr, h256, kE, 256, 256, 256, 256, 256, 0, 0, 0, 1, 1, 0, 1.0f);
      gemm(h256, wt_int2, int_b2, nullptr, comb_h, kE, kD, 256, 256, 256, kD, 0, 0, 0, 1, 0, 0, 1.0f);
      // messages
      k_msgin<<<ew_blocks((long long)kE * 384), 256, 0, stream>>>(node_h, from_idx, to_idx, comb_h, msgin);
      gemm(msgin, wt_msg1, msg_b1, nullptr, h256, kE, 256, 384, 384, 384, 256, 0, 0, 0, 1, 1, 0, 1.0f);
      gemm(h256, wt_msg2, msg_b2, msgs, nullptr, kE, kD, 256, 256, 256, kD, 0, 0, 0, 1, 0, 0, 1.0f);
      // agg = segment_sum(msgs, to_idx)
      k_zero<<<ew_blocks((long long)kN * kD), 256, 0, stream>>>(agg, (long long)kN * kD);
      k_scatter<<<ew_blocks((long long)kE * kD), 256, 0, stream>>>(msgs, to_idx, agg);
      // node update
      k_updin<<<ew_blocks((long long)kN * 256), 256, 0, stream>>>(node_h, agg, updin);
      gemm(updin, wt_upd1, upd_b1, nullptr, h256, kN, 256, 256, 256, 256, 256, 0, 0, 0, 1, 1, 0, 1.0f);
      gemm(h256, wt_upd2, upd_b2, nullptr, node_h, kN, kD, 256, 256, 256, kD, 0, 0, 0, 1, 0, 0, 1.0f);
      // edge update with new node states
      k_msgin<<<ew_blocks((long long)kE * 384), 256, 0, stream>>>(node_h, from_idx, to_idx, comb_h, msgin);
      gemm(msgin, wt_msg1, msg_b1, nullptr, h256, kE, 256, 384, 384, 384, 256, 0, 0, 0, 1, 1, 0, 1.0f);
      gemm(h256, wt_msg2, msg_b2, nullptr, edge_h, kE, kD, 256, 256, 256, kD, 0, 0, 0, 1, 0, 0, 1.0f);
    } else {
      // final: qcf = P @ sc (f32, reuse la), score = -sum relu(sq - qcf)
      gemm(P_h, epT + (size_t)kD * kM, nullptr, la, nullptr, kM, kD, kM,
           kM, kM, kD, (long long)kM * kM, 2LL * kD * kM, (long long)kM * kM, kB, 0, 0, 1.0f);
      k_score<<<kB, 256, 0, stream>>>(ep, la, out);
    }
  }
}
